// GATHeadLayer_32418413150992
// MI455X (gfx1250) — matrix operand order repaired and checked by reading, not compiled, not run on hardware
//
#include <hip/hip_runtime.h>
#include <hip/hip_bf16.h>

#define IN_DIM 256
#define HEAD_DIM 64
#define NEG_SLOPE 0.01f

typedef __attribute__((ext_vector_type(16))) _Float16 v16h;
typedef __attribute__((ext_vector_type(8)))  float    v8f;

// -------- order-preserving float<->uint encoding for atomicMax on floats ----
__device__ __forceinline__ unsigned fenc(float f) {
    unsigned u = __float_as_uint(f);
    return (u & 0x80000000u) ? ~u : (u | 0x80000000u);
}
__device__ __forceinline__ float fdec(unsigned u) {
    unsigned v = (u & 0x80000000u) ? (u & 0x7FFFFFFFu) : ~u;
    return __uint_as_float(v);
}

// -------- 1) init: zero h_out accumulator, denom, encoded max ---------------
__global__ __launch_bounds__(256) void k_init(float* __restrict__ hout,
                                              float* __restrict__ denom,
                                              unsigned* __restrict__ menc,
                                              int n_nodes) {
    int i = blockIdx.x * 256 + threadIdx.x;
    int total = n_nodes * HEAD_DIM;
    if (i < total) hout[i] = 0.0f;
    if (i < n_nodes) { denom[i] = 0.0f; menc[i] = 0u; }
}

// -------- 2) z = h @ fc_w^T via WMMA f32_16x16x32_f16 -----------------------
// Block = 256 threads = 8 waves. Each wave computes a 16x64 slab of z.
// fc_w is staged to LDS as f16 (64x256 = 32KB).
__global__ __launch_bounds__(256) void k_gemm_wmma(const float* __restrict__ h,
                                                   const float* __restrict__ fcw,
                                                   float* __restrict__ z,
                                                   int n_nodes) {
    __shared__ _Float16 bsh[HEAD_DIM * IN_DIM];   // 32 KB
    const int tid = threadIdx.x;
    #pragma unroll 4
    for (int i = tid; i < HEAD_DIM * IN_DIM; i += 256)
        bsh[i] = (_Float16)fcw[i];
    __syncthreads();

    const int wave   = tid >> 5;
    const int lane   = tid & 31;
    const int lhalf  = lane >> 4;     // 0 or 1
    const int l16    = lane & 15;
    const int m_base = (blockIdx.x * 8 + wave) * 16;
    if (m_base >= n_nodes) return;    // after the only barrier: safe

    // clamp row for OOB-safe loads (writes are guarded separately)
    int m = m_base + l16;
    if (m >= n_nodes) m = n_nodes - 1;
    const float* arow = h + (size_t)m * IN_DIM;

    v8f acc[4];
    #pragma unroll
    for (int t = 0; t < 4; ++t) acc[t] = (v8f){0.f,0.f,0.f,0.f,0.f,0.f,0.f,0.f};

    #pragma unroll
    for (int kk = 0; kk < IN_DIM; kk += 32) {
        // A fragment (16x32 f16): lanes 0-15 rows M=0..15 K=kk+0..15;
        // lanes 16-31 same rows, K=kk+16..31.
        const int kb = kk + lhalf * 16;
        float4 a0 = *(const float4*)(arow + kb);
        float4 a1 = *(const float4*)(arow + kb + 4);
        float4 a2 = *(const float4*)(arow + kb + 8);
        float4 a3 = *(const float4*)(arow + kb + 12);
        v16h A;
        A[0]=(_Float16)a0.x; A[1]=(_Float16)a0.y; A[2]=(_Float16)a0.z; A[3]=(_Float16)a0.w;
        A[4]=(_Float16)a1.x; A[5]=(_Float16)a1.y; A[6]=(_Float16)a1.z; A[7]=(_Float16)a1.w;
        A[8]=(_Float16)a2.x; A[9]=(_Float16)a2.y; A[10]=(_Float16)a2.z; A[11]=(_Float16)a2.w;
        A[12]=(_Float16)a3.x; A[13]=(_Float16)a3.y; A[14]=(_Float16)a3.z; A[15]=(_Float16)a3.w;

        // B fragments (32x16 f16): col n = l16 (+16 for upper half lanes'
        // K-shift): lanes 0-15 K=kk..kk+15, lanes 16-31 K=kk+16..kk+31.
        const int kb2 = kk + lhalf * 16;
        #pragma unroll
        for (int t = 0; t < 4; ++t) {
            const int n = t * 16 + l16;
            const v16h B = *(const v16h*)(bsh + n * IN_DIM + kb2);
            acc[t] = __builtin_amdgcn_wmma_f32_16x16x32_f16(
                false, A, false, B, (short)0, acc[t], false, false);
        }
    }

    // C/D layout: VGPR r, lanes 0-15 -> row m_base+r, col l16; lanes 16-31 -> row m_base+r+8
    #pragma unroll
    for (int t = 0; t < 4; ++t) {
        #pragma unroll
        for (int r = 0; r < 8; ++r) {
            const int row = m_base + r + lhalf * 8;
            const int col = t * 16 + l16;
            if (row < n_nodes)
                z[(size_t)row * HEAD_DIM + col] = acc[t][r];
        }
    }
}

// -------- 3) el/er = z @ w_l, z @ w_r : one wave per node -------------------
__global__ __launch_bounds__(256) void k_elr(const float* __restrict__ z,
                                             const float* __restrict__ attn_w,
                                             float* __restrict__ el,
                                             float* __restrict__ er,
                                             int n_nodes) {
    const int gwave = (blockIdx.x * 256 + threadIdx.x) >> 5;
    const int lane  = threadIdx.x & 31;
    if (gwave >= n_nodes) return;
    const float* zr = z + (size_t)gwave * HEAD_DIM;
    const float z0 = zr[lane], z1 = zr[lane + 32];
    float l = z0 * attn_w[lane]      + z1 * attn_w[lane + 32];
    float r = z0 * attn_w[64 + lane] + z1 * attn_w[96 + lane];
    #pragma unroll
    for (int off = 16; off >= 1; off >>= 1) {
        l += __shfl_xor(l, off, 32);
        r += __shfl_xor(r, off, 32);
    }
    if (lane == 0) { el[gwave] = l; er[gwave] = r; }
}

// -------- 4) per-edge leaky-relu + segment max (encoded atomicMax) ----------
__global__ __launch_bounds__(256) void k_edge_max(const int* __restrict__ src,
                                                  const int* __restrict__ dst,
                                                  const float* __restrict__ el,
                                                  const float* __restrict__ er,
                                                  float* __restrict__ ebuf,
                                                  unsigned* __restrict__ menc,
                                                  int n_edges) {
    const int i = blockIdx.x * 256 + threadIdx.x;
    if (i >= n_edges) return;
    const int s = src[i], d = dst[i];
    float e = el[s] + er[d];
    e = (e > 0.0f) ? e : e * NEG_SLOPE;
    ebuf[i] = e;
    atomicMax(menc + d, fenc(e));
}

// -------- 5) per-edge exp + segment sum -------------------------------------
__global__ __launch_bounds__(256) void k_edge_exp(const int* __restrict__ dst,
                                                  float* __restrict__ ebuf,
                                                  const unsigned* __restrict__ menc,
                                                  float* __restrict__ denom,
                                                  int n_edges) {
    const int i = blockIdx.x * 256 + threadIdx.x;
    if (i >= n_edges) return;
    const int d = dst[i];
    const float ex = __expf(ebuf[i] - fdec(menc[d]));
    ebuf[i] = ex;
    atomicAdd(denom + d, ex);
}

// -------- 6) weighted scatter-add: 64 threads per edge ----------------------
__global__ __launch_bounds__(256) void k_scatter(const int* __restrict__ src,
                                                 const int* __restrict__ dst,
                                                 const float* __restrict__ ebuf,
                                                 const float* __restrict__ denom,
                                                 const float* __restrict__ z,
                                                 float* __restrict__ hout,
                                                 int n_edges) {
    const unsigned gid  = blockIdx.x * 256u + threadIdx.x;
    const unsigned edge = gid >> 6;
    const unsigned k    = gid & 63u;
    if (edge >= (unsigned)n_edges) return;
    const int s = src[edge], d = dst[edge];
    const float alpha = ebuf[edge] / denom[d];
    atomicAdd(hout + (size_t)d * HEAD_DIM + k,
              alpha * z[(size_t)s * HEAD_DIM + k]);
}

// -------- 7) finalize: relu(h_out * snorm) ----------------------------------
__global__ __launch_bounds__(256) void k_finalize(float* __restrict__ hout,
                                                  const float* __restrict__ snorm,
                                                  int n_nodes) {
    const int i = blockIdx.x * 256 + threadIdx.x;
    if (i >= n_nodes * HEAD_DIM) return;
    const float v = hout[i] * snorm[i >> 6];
    hout[i] = (v > 0.0f) ? v : 0.0f;
}

extern "C" void kernel_launch(void* const* d_in, const int* in_sizes, int n_in,
                              void* d_out, int out_size, void* d_ws, size_t ws_size,
                              hipStream_t stream) {
    const float* h      = (const float*)d_in[0];
    const float* snorm  = (const float*)d_in[1];
    const float* fcw    = (const float*)d_in[2];
    const float* attn_w = (const float*)d_in[3];
    const int*   src    = (const int*)d_in[4];
    const int*   dst    = (const int*)d_in[5];
    float*       hout   = (float*)d_out;

    const int N = in_sizes[0] / IN_DIM;   // nodes
    const int E = in_sizes[4];            // edges

    // workspace carve-out
    char* ws = (char*)d_ws;
    size_t off = 0;
    float* z      = (float*)(ws + off); off += (size_t)N * HEAD_DIM * sizeof(float);
    float* el     = (float*)(ws + off); off += (size_t)N * sizeof(float);
    float* er     = (float*)(ws + off); off += (size_t)N * sizeof(float);
    unsigned* men = (unsigned*)(ws + off); off += (size_t)N * sizeof(unsigned);
    float* denom  = (float*)(ws + off); off += (size_t)N * sizeof(float);
    float* ebuf   = (float*)(ws + off); off += (size_t)E * sizeof(float);
    (void)ws_size; (void)n_in; (void)out_size;

    const int bInit  = (N * HEAD_DIM + 255) / 256;
    k_init<<<bInit, 256, 0, stream>>>(hout, denom, men, N);

    const int rowTiles = (N + 15) / 16;
    const int bGemm    = (rowTiles + 7) / 8;
    k_gemm_wmma<<<bGemm, 256, 0, stream>>>(h, fcw, z, N);

    const int bElr = (N + 7) / 8;              // 8 waves per block, wave/node
    k_elr<<<bElr, 256, 0, stream>>>(z, attn_w, el, er, N);

    const int bEdge = (E + 255) / 256;
    k_edge_max<<<bEdge, 256, 0, stream>>>(src, dst, el, er, ebuf, men, E);
    k_edge_exp<<<bEdge, 256, 0, stream>>>(dst, ebuf, men, denom, E);

    const long long scatterThreads = (long long)E * HEAD_DIM;
    const int bScatter = (int)((scatterThreads + 255) / 256);
    k_scatter<<<bScatter, 256, 0, stream>>>(src, dst, ebuf, denom, z, hout, E);

    k_finalize<<<bInit, 256, 0, stream>>>(hout, snorm, N);
}
